// MatchLSTM_26439818674453
// MI455X (gfx1250) — matrix operand order repaired and checked by reading, not compiled, not run on hardware
//
#include <hip/hip_runtime.h>
#include <hip/hip_bf16.h>
#include <stdint.h>

// MatchLSTM for MI455X (gfx1250, wave32, WMMA + TDM).
// All matmuls via v_wmma_f32_16x16x32_f16 (f16 operands, f32 accumulation).
// Sequential recurrences run as persistent single-workgroup kernels with the
// recurrent weight DMA'd into the 320KB WGP LDS by the Tensor Data Mover.

typedef __attribute__((ext_vector_type(16))) _Float16 v16h;
typedef __attribute__((ext_vector_type(8)))  _Float16 h8;
typedef __attribute__((ext_vector_type(8)))  float    v8f;

#define HH   150   // hidden
#define H3   450   // 3*hidden
#define DIM  300   // embedding dim
#define BB   8
#define TT   400
#define JJ   64
#define KPD  320   // K-pad for d=300 (multiple of 32)
#define KPH  160   // K-pad for h=150
#define NP3  464   // N-pad for 3h=450 (multiple of 16)
#define NP1  160   // N-pad for h=150

// ---- WMMA fragment loaders (wave32 layouts per CDNA5 ISA 7.12.2) ----------
// A 16x32 f16: lane L holds row M=L%16; khalf=L/16; elems 0..7 -> K=kh*8+e,
// elems 8..15 -> K=16+kh*8+(e-8). Two contiguous 16B loads per lane.
__device__ __forceinline__ v16h frag_a(const _Float16* A, int lda, int mBase,
                                       int k0, int lane) {
  int m  = mBase + (lane & 15);
  int kh = (lane >> 4) & 1;
  const _Float16* p = A + (size_t)m * lda + k0 + kh * 8;
  h8 lo = *(const h8*)p;
  h8 hi = *(const h8*)(p + 16);
  v16h r;
#pragma unroll
  for (int i = 0; i < 8; ++i) { r[i] = lo[i]; r[i + 8] = hi[i]; }
  return r;
}

// B 32x16 f16 (stored as Bmat[N][K] row-major): lane L holds col N=L%16;
// khalf=L/16; elems 0..7 -> K=kh*8+e, elems 8..15 -> K=16+kh*8+(e-8).
__device__ __forceinline__ v16h frag_b(const _Float16* B, int ldb, int nBase,
                                       int k0, int lane) {
  int n  = nBase + (lane & 15);
  int kh = (lane >> 4) & 1;
  const _Float16* p = B + (size_t)n * ldb + k0 + kh * 8;
  h8 lo = *(const h8*)p;
  h8 hi = *(const h8*)(p + 16);
  v16h r;
#pragma unroll
  for (int i = 0; i < 8; ++i) { r[i] = lo[i]; r[i + 8] = hi[i]; }
  return r;
}

__device__ __forceinline__ float sigmoidf_(float x) {
  return 1.f / (1.f + __expf(-x));
}

// ---- Tensor Data Mover: DMA `bytes` (multiple of 8) global -> LDS ---------
// D# per CDNA5 ISA 8.3/8.4: count=1, type=2, data_size=8B, 2D tile of
// (bytes/8) x 1 elements; contiguous global -> contiguous LDS.
// Issued by one wave; completion via s_wait_tensorcnt in that wave.
#if __has_builtin(__builtin_amdgcn_tensor_load_to_lds)
#define HAVE_TDM 1
__device__ __forceinline__ void tdm_load_to_lds(const void* gptr, void* lptr,
                                                uint32_t bytes) {
  typedef unsigned int u32x4 __attribute__((ext_vector_type(4)));
  typedef int i32x4 __attribute__((ext_vector_type(4)));
  typedef int i32x8 __attribute__((ext_vector_type(8)));
  uint64_t ga = (uint64_t)(uintptr_t)gptr;
  uint32_t lds = (uint32_t)(uintptr_t)lptr;  // low 32 bits = LDS byte offset
  uint32_t n8 = bytes >> 3;                  // elements of 8 bytes
  u32x4 g0;
  g0[0] = 1u;                                      // count=1, user D#
  g0[1] = lds;                                     // lds_addr [63:32]
  g0[2] = (uint32_t)ga;                            // global_addr [95:64]
  g0[3] = (uint32_t)(ga >> 32) | 0x80000000u;      // addr[56:32] | type=2
  i32x8 g1;
  g1[0] = (int)(3u << 16);          // workgroup_mask=0, data_size=8B
  g1[1] = (int)(n8 << 16);          // tensor_dim0[15:0] @ bit48
  g1[2] = (int)((n8 >> 16) | (1u << 16));  // tensor_dim0 hi | tensor_dim1=1
  g1[3] = (int)(n8 << 16);          // tile_dim0 @ bit112 (tensor_dim1 hi = 0)
  g1[4] = 1;                        // tile_dim1=1, tile_dim2=0
  g1[5] = (int)n8;                  // tensor_dim0_stride @ bit160
  g1[6] = 0;
  g1[7] = 0;
  i32x4 z4 = {0, 0, 0, 0};
#if defined(__clang_major__) && __clang_major__ >= 23
  i32x8 z8 = {0, 0, 0, 0, 0, 0, 0, 0};
  __builtin_amdgcn_tensor_load_to_lds(g0, g1, z4, z4, z8, 0);
#else
  __builtin_amdgcn_tensor_load_to_lds(g0, g1, z4, z4, 0);
#endif
}
#else
#define HAVE_TDM 0
#endif

// Stage `bytes` from global into LDS; call from all threads, no barrier inside (caller syncs).
__device__ __forceinline__ void stage_weights(const void* g, void* l,
                                              uint32_t bytes, int tid,
                                              int nthr) {
#if HAVE_TDM
  if (tid == 0) {
    tdm_load_to_lds(g, l, bytes);
    __builtin_amdgcn_s_wait_tensorcnt(0);
  }
#else
  const uint4* gs = (const uint4*)g;
  uint4* ls = (uint4*)l;
  for (uint32_t i = tid; i < bytes / 16; i += nthr) ls[i] = gs[i];
#endif
}

// ---- weight pack: f32 -> padded f16 [Npad][Kpad], optional transpose ------
__global__ void pack_f16_kernel(const float* __restrict__ src,
                                _Float16* __restrict__ dst, int srcLd,
                                int colOff, int rowsValid, int colsValid,
                                int Npad, int Kpad, int transposed) {
  int idx = blockIdx.x * blockDim.x + threadIdx.x;
  if (idx >= Npad * Kpad) return;
  int n = idx / Kpad, k = idx - n * Kpad;
  float v = 0.f;
  if (n < rowsValid && k < colsValid)
    v = transposed ? src[(size_t)k * srcLd + n + colOff]
                   : src[(size_t)n * srcLd + k + colOff];
  dst[idx] = (_Float16)v;
}

// ---- embedding gather -> padded f16 rows ---------------------------------
__global__ void gather_embed_kernel(const int* __restrict__ tok,
                                    const float* __restrict__ E,
                                    _Float16* __restrict__ dst, int d,
                                    int Kpad) {
  int row = blockIdx.x;
  const float* e = E + (size_t)tok[row] * d;
  for (int c = threadIdx.x; c < Kpad; c += blockDim.x)
    dst[(size_t)row * Kpad + c] = (c < d) ? (_Float16)e[c] : (_Float16)0.f;
}

// ---- generic WMMA GEMM: C[M][ldc] = A[M][lda] @ Bmat[N][K]^T (+bias) ------
__global__ __launch_bounds__(128) void gemm_wmma_kernel(
    const _Float16* __restrict__ A, int lda, const _Float16* __restrict__ B,
    int ldb, float* __restrict__ C, int ldc, const float* __restrict__ bias,
    int Nreal, int Kpad, int tilesM, int tilesN) {
  int wave = (blockIdx.x * blockDim.x + threadIdx.x) >> 5;
  int lane = threadIdx.x & 31;
  int tm = wave / tilesN;
  int tn = wave - tm * tilesN;
  if (tm >= tilesM) return;
  v8f acc = {};
  for (int k0 = 0; k0 < Kpad; k0 += 32) {
    v16h a = frag_a(A, lda, tm * 16, k0, lane);
    v16h b = frag_b(B, ldb, tn * 16, k0, lane);
    acc = __builtin_amdgcn_wmma_f32_16x16x32_f16(false, a, false, b, (short)0,
                                                 acc, false, false);
  }
  int n  = tn * 16 + (lane & 15);
  int kh = (lane >> 4) & 1;
  float bv = (bias != nullptr && n < Nreal) ? bias[n] : 0.f;
#pragma unroll
  for (int v = 0; v < 8; ++v)
    C[(size_t)(tm * 16 + v + 8 * kh) * ldc + n] = acc[v] + bv;
}

// ---- persistent GRU scan: one workgroup, Whh resident in LDS --------------
__global__ __launch_bounds__(512) void gru_scan_kernel(
    const float* __restrict__ xp /*[B*T][NP3] incl bih*/,
    const _Float16* __restrict__ Whh /*[NP3][KPH]*/,
    const float* __restrict__ bhh, _Float16* __restrict__ Hout /*[B*T][KPH]*/,
    int Tlen, int Mrows) {
  extern __shared__ char smem[];
  _Float16* sWhh = (_Float16*)smem;                              // NP3*KPH f16
  _Float16* sh   = (_Float16*)(smem + NP3 * KPH * 2);            // 16*KPH f16
  float* shf = (float*)(smem + NP3 * KPH * 2 + 16 * KPH * 2);    // 16*KPH f32
  float* shp =
      (float*)(smem + NP3 * KPH * 2 + 16 * KPH * 2 + 16 * KPH * 4); // 16*NP3
  const int tid = threadIdx.x, nthr = blockDim.x;
  // TDM: DMA recurrent weight into LDS (145KB; fits 320KB WGP LDS)
  stage_weights(Whh, sWhh, NP3 * KPH * 2, tid, nthr);
  for (int i = tid; i < 16 * KPH / 2; i += nthr) ((uint32_t*)sh)[i] = 0u;
  for (int i = tid; i < 16 * KPH; i += nthr) shf[i] = 0.f;
  // zero Hout once so padded K columns (150..159) are valid GEMM input
  for (int i = tid; i < Mrows * KPH / 2; i += nthr) ((uint32_t*)Hout)[i] = 0u;
  __syncthreads();
  const int wave = tid >> 5, lane = tid & 31;
  for (int t = 0; t < Tlen; ++t) {
    // hp[16,464] = h_f16[16,160] @ Whh^T  (29 tiles over 16 waves)
    for (int tile = wave; tile < NP3 / 16; tile += 16) {
      v8f acc = {};
#pragma unroll
      for (int kk = 0; kk < KPH / 32; ++kk) {
        v16h a = frag_a(sh, KPH, 0, kk * 32, lane);
        v16h b = frag_b(sWhh, KPH, tile * 16, kk * 32, lane);
        acc = __builtin_amdgcn_wmma_f32_16x16x32_f16(false, a, false, b,
                                                     (short)0, acc, false,
                                                     false);
      }
      int n  = tile * 16 + (lane & 15);
      int kh = (lane >> 4) & 1;
      float bv = (n < H3) ? bhh[n] : 0.f;
#pragma unroll
      for (int v = 0; v < 8; ++v) shp[(v + 8 * kh) * NP3 + n] = acc[v] + bv;
    }
    __syncthreads();
    // GRU cell (r,z,n gate order) on 8*150 lanes
    for (int i = tid; i < BB * HH; i += nthr) {
      int b = i / HH, f = i - b * HH;
      const float* x = xp + (size_t)(b * Tlen + t) * NP3;
      float hr = shp[b * NP3 + f];
      float hz = shp[b * NP3 + HH + f];
      float hn = shp[b * NP3 + 2 * HH + f];
      float r  = sigmoidf_(x[f] + hr);
      float z  = sigmoidf_(x[HH + f] + hz);
      float nn = tanhf(x[2 * HH + f] + r * hn);
      float hnew = (1.f - z) * nn + z * shf[b * KPH + f];
      shf[b * KPH + f] = hnew;
      sh[b * KPH + f]  = (_Float16)hnew;
      Hout[(size_t)(b * Tlen + t) * KPH + f] = (_Float16)hnew;
    }
    __syncthreads();
  }
}

// ---- persistent match recurrence ------------------------------------------
__global__ __launch_bounds__(512) void match_scan_kernel(
    const float* __restrict__ whq /*[B*J][KPH]*/,
    const float* __restrict__ whp /*[B*T][KPH]*/,
    const float* __restrict__ zx /*[B*T][NP3]*/,
    const _Float16* __restrict__ Hq /*[B*J][KPH]*/,
    const float* __restrict__ wvec /*[150]*/,
    const _Float16* __restrict__ WrT /*[NP1][KPH]*/,
    const _Float16* __restrict__ mWihB /*[NP3][KPH]*/,
    const _Float16* __restrict__ mWhh /*[NP3][KPH]*/,
    const float* __restrict__ m_bih, const float* __restrict__ m_bhh,
    float* __restrict__ out /*[B][T+1][HH]*/, int Tlen) {
  extern __shared__ char smem[];
  char* p = smem;
  _Float16* sWhh = (_Float16*)p; p += NP3 * KPH * 2;  // m_Whh cache
  _Float16* shid = (_Float16*)p; p += 16 * KPH * 2;   // hidden f16
  _Float16* saq  = (_Float16*)p; p += 16 * KPH * 2;   // attn_query f16
  float* shidf = (float*)p;      p += 16 * KPH * 4;   // hidden f32
  float* swhr  = (float*)p;      p += 16 * KPH * 4;   // hidden @ Wr
  float* shp   = (float*)p;      p += 16 * NP3 * 4;   // hidden @ m_Whh^T
  float* sxp   = (float*)p;      p += 16 * NP3 * 4;   // aq @ m_WihB^T + bih
  float* sattn = (float*)p;      p += BB * JJ * 4;    // attention scores
  const int tid = threadIdx.x, nthr = blockDim.x;
  stage_weights(mWhh, sWhh, NP3 * KPH * 2, tid, nthr);
  for (int i = tid; i < 16 * KPH / 2; i += nthr) {
    ((uint32_t*)shid)[i] = 0u;
    ((uint32_t*)saq)[i]  = 0u;
  }
  for (int i = tid; i < 16 * KPH; i += nthr) shidf[i] = 0.f;
  for (int i = tid; i < BB * HH; i += nthr)  // H_r[:,0,:] = h0 = 0
    out[(size_t)(i / HH) * (TT + 1) * HH + (i % HH)] = 0.f;
  __syncthreads();
  const int wave = tid >> 5, lane = tid & 31;
  for (int t = 0; t < Tlen; ++t) {
    // 1) wh_r = hidden @ Wr  (10 N-tiles -> waves 0..9)
    if (wave < NP1 / 16) {
      v8f acc = {};
#pragma unroll
      for (int kk = 0; kk < KPH / 32; ++kk) {
        v16h a = frag_a(shid, KPH, 0, kk * 32, lane);
        v16h b = frag_b(WrT, KPH, wave * 16, kk * 32, lane);
        acc = __builtin_amdgcn_wmma_f32_16x16x32_f16(false, a, false, b,
                                                     (short)0, acc, false,
                                                     false);
      }
      int n  = wave * 16 + (lane & 15);
      int kh = (lane >> 4) & 1;
#pragma unroll
      for (int v = 0; v < 8; ++v) swhr[(v + 8 * kh) * KPH + n] = acc[v];
    }
    __syncthreads();
    // 2) attention scores (no softmax, per reference): 512 (b,j) pairs
    {
      int b = tid >> 6, j = tid & 63;
      const float* wq = whq + (size_t)(b * JJ + j) * KPH;
      const float* wp = whp + (size_t)(b * TT + t) * KPH;
      const float* wr = swhr + b * KPH;
      if (t + 1 < Tlen) __builtin_prefetch(wp + KPH, 0, 3);  // next timestep
      float s = 0.f;
      for (int f = 0; f < HH; ++f) s += tanhf(wq[f] + wp[f] + wr[f]) * wvec[f];
      sattn[tid] = s;
    }
    __syncthreads();
    // 3) attn_query[b,f] = sum_j attn[b,j] * Hq[b,j,f]
    for (int i = tid; i < BB * HH; i += nthr) {
      int b = i / HH, f = i - b * HH;
      float a = 0.f;
      for (int j = 0; j < JJ; ++j)
        a += sattn[b * JJ + j] * (float)Hq[(size_t)(b * JJ + j) * KPH + f];
      saq[b * KPH + f] = (_Float16)a;
    }
    __syncthreads();
    // 4) two WMMA GEMMs (58 tiles over 16 waves):
    //    sxp = aq @ m_Wih[:,h:]^T + m_bih ; shp = hidden @ m_Whh^T + m_bhh
    for (int g = wave; g < 2 * (NP3 / 16); g += 16) {
      bool first = g < NP3 / 16;
      int tile = first ? g : g - NP3 / 16;
      const _Float16* Am = first ? saq : shid;
      const _Float16* Bm = first ? mWihB : sWhh;
      const float* bias  = first ? m_bih : m_bhh;
      float* dst         = first ? sxp : shp;
      v8f acc = {};
#pragma unroll
      for (int kk = 0; kk < KPH / 32; ++kk) {
        v16h a = frag_a(Am, KPH, 0, kk * 32, lane);
        v16h b = frag_b(Bm, KPH, tile * 16, kk * 32, lane);
        acc = __builtin_amdgcn_wmma_f32_16x16x32_f16(false, a, false, b,
                                                     (short)0, acc, false,
                                                     false);
      }
      int n  = tile * 16 + (lane & 15);
      int kh = (lane >> 4) & 1;
      float bv = (n < H3) ? bias[n] : 0.f;
#pragma unroll
      for (int v = 0; v < 8; ++v) dst[(v + 8 * kh) * NP3 + n] = acc[v] + bv;
    }
    __syncthreads();
    // 5) GRU cell + emit hidden
    for (int i = tid; i < BB * HH; i += nthr) {
      int b = i / HH, f = i - b * HH;
      const float* z = zx + (size_t)(b * TT + t) * NP3;
      if (t + 1 < Tlen) __builtin_prefetch(z + NP3, 0, 3);  // next timestep
      float xr = z[f] + sxp[b * NP3 + f];
      float xz = z[HH + f] + sxp[b * NP3 + HH + f];
      float xn = z[2 * HH + f] + sxp[b * NP3 + 2 * HH + f];
      float hr = shp[b * NP3 + f];
      float hz = shp[b * NP3 + HH + f];
      float hn = shp[b * NP3 + 2 * HH + f];
      float r  = sigmoidf_(xr + hr);
      float zg = sigmoidf_(xz + hz);
      float nn = tanhf(xn + r * hn);
      float hnew = (1.f - zg) * nn + zg * shidf[b * KPH + f];
      shidf[b * KPH + f] = hnew;
      shid[b * KPH + f]  = (_Float16)hnew;
      out[((size_t)b * (TT + 1) + (t + 1)) * HH + f] = hnew;
    }
    __syncthreads();
  }
}

extern "C" void kernel_launch(void* const* d_in, const int* in_sizes, int n_in,
                              void* d_out, int out_size, void* d_ws,
                              size_t ws_size, hipStream_t stream) {
  (void)in_sizes; (void)n_in; (void)out_size; (void)ws_size;
  const int*   context = (const int*)d_in[0];
  const int*   query   = (const int*)d_in[1];
  const float* E       = (const float*)d_in[2];
  const float* Wq      = (const float*)d_in[3];
  const float* Wp      = (const float*)d_in[4];
  const float* Wr      = (const float*)d_in[5];
  const float* wvec    = (const float*)d_in[6];
  const float* ctx_Wih = (const float*)d_in[7];
  const float* ctx_Whh = (const float*)d_in[8];
  const float* ctx_bih = (const float*)d_in[9];
  const float* ctx_bhh = (const float*)d_in[10];
  const float* q_Wih   = (const float*)d_in[11];
  const float* q_Whh   = (const float*)d_in[12];
  const float* q_bih   = (const float*)d_in[13];
  const float* q_bhh   = (const float*)d_in[14];
  const float* m_Wih   = (const float*)d_in[15];
  const float* m_Whh   = (const float*)d_in[16];
  const float* m_bih   = (const float*)d_in[17];
  const float* m_bhh   = (const float*)d_in[18];
  float* out = (float*)d_out;

  char* ws = (char*)d_ws;
  size_t off = 0;
  auto carve = [&](size_t bytes) {
    char* p = ws + off;
    off += (bytes + 255) & ~(size_t)255;
    return p;
  };
  _Float16* ecF   = (_Float16*)carve((size_t)BB * TT * KPD * 2);
  _Float16* eqF   = (_Float16*)carve((size_t)BB * JJ * KPD * 2);
  _Float16* HcF   = (_Float16*)carve((size_t)BB * TT * KPH * 2);
  _Float16* HqF   = (_Float16*)carve((size_t)BB * JJ * KPH * 2);
  _Float16* WihC  = (_Float16*)carve((size_t)NP3 * KPD * 2);
  _Float16* WihQ  = (_Float16*)carve((size_t)NP3 * KPD * 2);
  _Float16* WhhC  = (_Float16*)carve((size_t)NP3 * KPH * 2);
  _Float16* WhhQ  = (_Float16*)carve((size_t)NP3 * KPH * 2);
  _Float16* mWhhF = (_Float16*)carve((size_t)NP3 * KPH * 2);
  _Float16* mWihA = (_Float16*)carve((size_t)NP3 * KPH * 2);
  _Float16* mWihB = (_Float16*)carve((size_t)NP3 * KPH * 2);
  _Float16* WqT   = (_Float16*)carve((size_t)NP1 * KPH * 2);
  _Float16* WpT   = (_Float16*)carve((size_t)NP1 * KPH * 2);
  _Float16* WrT   = (_Float16*)carve((size_t)NP1 * KPH * 2);
  float* xpC = (float*)carve((size_t)BB * TT * NP3 * 4);
  float* xpQ = (float*)carve((size_t)BB * JJ * NP3 * 4);
  float* whp = (float*)carve((size_t)BB * TT * KPH * 4);
  float* whq = (float*)carve((size_t)BB * JJ * KPH * 4);
  float* zx  = (float*)carve((size_t)BB * TT * NP3 * 4);

  // ---- pack weights to padded f16 (B-matrix [N][K] layout) ----
  auto pack = [&](const float* src, _Float16* dst, int srcLd, int colOff,
                  int rows, int cols, int Npad, int Kpad, int tr) {
    int n = Npad * Kpad;
    pack_f16_kernel<<<(n + 255) / 256, 256, 0, stream>>>(
        src, dst, srcLd, colOff, rows, cols, Npad, Kpad, tr);
  };
  pack(ctx_Wih, WihC, DIM, 0, H3, DIM, NP3, KPD, 0);
  pack(q_Wih,   WihQ, DIM, 0, H3, DIM, NP3, KPD, 0);
  pack(ctx_Whh, WhhC, HH, 0, H3, HH, NP3, KPH, 0);
  pack(q_Whh,   WhhQ, HH, 0, H3, HH, NP3, KPH, 0);
  pack(m_Whh,   mWhhF, HH, 0, H3, HH, NP3, KPH, 0);
  pack(m_Wih,   mWihA, 2 * HH, 0,  H3, HH, NP3, KPH, 0);  // m_Wih[:, :h]
  pack(m_Wih,   mWihB, 2 * HH, HH, H3, HH, NP3, KPH, 0);  // m_Wih[:, h:]
  pack(Wq, WqT, HH, 0, HH, HH, NP1, KPH, 1);  // transpose: Bmat[n][k]=Wq[k][n]
  pack(Wp, WpT, HH, 0, HH, HH, NP1, KPH, 1);
  pack(Wr, WrT, HH, 0, HH, HH, NP1, KPH, 1);

  // ---- embedding gather ----
  gather_embed_kernel<<<BB * TT, 128, 0, stream>>>(context, E, ecF, DIM, KPD);
  gather_embed_kernel<<<BB * JJ, 128, 0, stream>>>(query, E, eqF, DIM, KPD);

  auto gemm = [&](const _Float16* A, int lda, const _Float16* B, int ldb,
                  float* C, int ldc, const float* bias, int Nreal, int Kpad,
                  int tilesM, int tilesN) {
    int waves = tilesM * tilesN;
    int blocks = (waves + 3) / 4;  // 4 waves per 128-thread block
    gemm_wmma_kernel<<<blocks, 128, 0, stream>>>(A, lda, B, ldb, C, ldc, bias,
                                                 Nreal, Kpad, tilesM, tilesN);
  };

  // input projections (include b_ih)
  gemm(ecF, KPD, WihC, KPD, xpC, NP3, ctx_bih, H3, KPD, (BB * TT) / 16,
       NP3 / 16);
  gemm(eqF, KPD, WihQ, KPD, xpQ, NP3, q_bih, H3, KPD, (BB * JJ) / 16,
       NP3 / 16);

  // encoder GRU scans (persistent workgroups, Whh in LDS via TDM)
  size_t lds1 =
      (size_t)NP3 * KPH * 2 + 16 * KPH * 2 + 16 * KPH * 4 + 16 * NP3 * 4;
  gru_scan_kernel<<<1, 512, lds1, stream>>>(xpC, WhhC, ctx_bhh, HcF, TT,
                                            BB * TT);
  gru_scan_kernel<<<1, 512, lds1, stream>>>(xpQ, WhhQ, q_bhh, HqF, JJ,
                                            BB * JJ);

  // precomputed match-step projections
  gemm(HcF, KPH, WpT, KPH, whp, KPH, nullptr, HH, KPH, (BB * TT) / 16,
       NP1 / 16);
  gemm(HqF, KPH, WqT, KPH, whq, KPH, nullptr, HH, KPH, (BB * JJ) / 16,
       NP1 / 16);
  gemm(HcF, KPH, mWihA, KPH, zx, NP3, nullptr, H3, KPH, (BB * TT) / 16,
       NP3 / 16);

  // match recurrence
  size_t lds2 = (size_t)NP3 * KPH * 2 + 2 * (16 * KPH * 2) +
                2 * (16 * KPH * 4) + 2 * (16 * NP3 * 4) + BB * JJ * 4;
  match_scan_kernel<<<1, 512, lds2, stream>>>(whq, whp, zx, HqF, wvec, WrT,
                                              mWihB, mWhhF, m_bih, m_bhh, out,
                                              TT);
}
